// MultiResolutionSSM_80564996538674
// MI455X (gfx1250) — matrix operand order repaired and checked by reading, not compiled, not run on hardware
//
#include <hip/hip_runtime.h>
#include <hip/hip_bf16.h>
#include <math.h>

// ---------------- types for WMMA ----------------
typedef __bf16 v8bf  __attribute__((ext_vector_type(8)));
typedef __bf16 v16bf __attribute__((ext_vector_type(16)));
typedef float  v8f   __attribute__((ext_vector_type(8)));

// ---------------- DWT filters (db4) ----------------
__constant__ float c_dec_lo[8] = {
    -0.010597401784997278f,  0.032883011666982945f,  0.030841381835986965f,
    -0.18703481171888114f,  -0.02798376941698385f,   0.6308807679295904f,
     0.7148465705525415f,    0.23037781330885523f};
__constant__ float c_dec_hi[8] = {
    -0.23037781330885523f,   0.7148465705525415f,   -0.6308807679295904f,
    -0.02798376941698385f,   0.18703481171888114f,   0.030841381835986965f,
    -0.032883011666982945f, -0.010597401784997278f};

// ============================================================
// DWT: single level, both lo & hi outputs in one pass.
// grid = B*outlen blocks, 512 threads (one per channel h).
// ============================================================
__global__ void dwt_kernel(const float* __restrict__ in,
                           float* __restrict__ outA, float* __restrict__ outD,
                           int T, int outlen) {
    int bo = blockIdx.x;
    int h  = threadIdx.x;               // H = 512
    int b  = bo / outlen;
    int t  = bo % outlen;
    const float* xb = in + (size_t)b * T * 512;
    float accA = 0.f, accD = 0.f;
#pragma unroll
    for (int k = 0; k < 8; ++k) {
        int idx = 1 + k + 2 * t - 7;
        if (idx < 0)  idx = -idx - 1;
        if (idx >= T) idx = 2 * T - 1 - idx;
        float v = xb[(size_t)idx * 512 + h];
        accA += c_dec_lo[7 - k] * v;
        accD += c_dec_hi[7 - k] * v;
    }
    size_t o = ((size_t)b * outlen + t) * 512 + h;
    outA[o] = accA;
    outD[o] = accD;
}

// ============================================================
// f32 -> bf16 conversion
// ============================================================
__global__ void f32_to_bf16_kernel(const float* __restrict__ in,
                                   __bf16* __restrict__ out, size_t n) {
    size_t i = (size_t)blockIdx.x * blockDim.x + threadIdx.x;
    if (i < n) out[i] = (__bf16)in[i];
}

// ============================================================
// bf16 WMMA GEMM with 2x2 register tiling:
//   C[M x N] = A[M x K] * B[N x K]^T   (f32 accumulate)
// K % 32 == 0, N % 32 == 0. 128 threads = 4 waves; each wave owns a
// 32x32 output tile = 2x2 sub-tiles of 16x16, so each k-step issues
// 4 x v_wmma_f32_16x16x32_bf16 for 8 x b128 loads (2 loads/WMMA).
// Fragment layouts per CDNA5 ISA 7.12.2:
//   A 16x32 bf16 : lanes 0-15 -> M=lane, elems 0..7=K(k0+0..7), 8..15=K(k0+16..23)
//                  lanes 16-31 -> M=lane-16, elems=K(k0+8..15), K(k0+24..31)
//   B 32x16 bf16 : lanes 0-15 -> N=lane, K=k0+0..7, k0+16..23 ; lanes 16-31 -> K=k0+8..15, k0+24..31
//   C/D 16x16 f32: VGPR r, lanes 0-15 -> M=r, lanes 16-31 -> M=r+8 ; N=lane&15
// Wave-uniform tail exit + row-clamped loads keep EXEC all-ones for WMMA;
// stores are bounds-checked per row.
// ============================================================
__global__ void gemm_bf16_wmma_kernel(const __bf16* __restrict__ A,
                                      const __bf16* __restrict__ Bm,
                                      float* __restrict__ C,
                                      int M, int N, int K) {
    const int lane  = threadIdx.x & 31;
    const int wave  = threadIdx.x >> 5;
    const int tileN = blockIdx.x * 32;
    const int tileM = (blockIdx.y * 4 + wave) * 32;
    if (tileM >= M) return;                 // wave-uniform: EXEC stays full

    const int hf  = lane >> 4;              // 0 or 1 (half-wave)
    const int l15 = lane & 15;

    int arow0 = tileM + l15;       if (arow0 >= M) arow0 = M - 1;
    int arow1 = tileM + 16 + l15;  if (arow1 >= M) arow1 = M - 1;
    const int aoff = hf * 8;                // A: K sub-offset per half-wave
    const int brow0 = tileN + l15;          // B rows = output columns
    const int brow1 = tileN + 16 + l15;
    const int boff = hf * 8;                // B: K sub-offset per half-wave

    const __bf16* ap0 = A  + (size_t)arow0 * K;
    const __bf16* ap1 = A  + (size_t)arow1 * K;
    const __bf16* bp0 = Bm + (size_t)brow0 * K;
    const __bf16* bp1 = Bm + (size_t)brow1 * K;

    v8f acc00 = {0.f,0.f,0.f,0.f,0.f,0.f,0.f,0.f};
    v8f acc01 = acc00, acc10 = acc00, acc11 = acc00;

    for (int k0 = 0; k0 < K; k0 += 32) {
        union { v16bf v; v8bf h[2]; } fa0, fa1, fb0, fb1;
        fa0.h[0] = *reinterpret_cast<const v8bf*>(ap0 + k0 + aoff);
        fa0.h[1] = *reinterpret_cast<const v8bf*>(ap0 + k0 + 16 + aoff);
        fa1.h[0] = *reinterpret_cast<const v8bf*>(ap1 + k0 + aoff);
        fa1.h[1] = *reinterpret_cast<const v8bf*>(ap1 + k0 + 16 + aoff);
        fb0.h[0] = *reinterpret_cast<const v8bf*>(bp0 + k0 + boff);
        fb0.h[1] = *reinterpret_cast<const v8bf*>(bp0 + k0 + 16 + boff);
        fb1.h[0] = *reinterpret_cast<const v8bf*>(bp1 + k0 + boff);
        fb1.h[1] = *reinterpret_cast<const v8bf*>(bp1 + k0 + 16 + boff);
        if (k0 + 32 < K) {                  // prefetch next K chunk
            __builtin_prefetch(ap0 + k0 + 32, 0, 3);
            __builtin_prefetch(ap1 + k0 + 32, 0, 3);
            __builtin_prefetch(bp0 + k0 + 32, 0, 3);
            __builtin_prefetch(bp1 + k0 + 32, 0, 3);
        }
        acc00 = __builtin_amdgcn_wmma_f32_16x16x32_bf16(
            false, fa0.v, false, fb0.v, (short)0, acc00, false, false);
        acc01 = __builtin_amdgcn_wmma_f32_16x16x32_bf16(
            false, fa0.v, false, fb1.v, (short)0, acc01, false, false);
        acc10 = __builtin_amdgcn_wmma_f32_16x16x32_bf16(
            false, fa1.v, false, fb0.v, (short)0, acc10, false, false);
        acc11 = __builtin_amdgcn_wmma_f32_16x16x32_bf16(
            false, fa1.v, false, fb1.v, (short)0, acc11, false, false);
    }

#pragma unroll
    for (int r = 0; r < 8; ++r) {
        int m0 = tileM + r + hf * 8;
        int m1 = m0 + 16;
        size_t c0 = (size_t)m0 * N + tileN + l15;
        size_t c1 = (size_t)m1 * N + tileN + l15;
        if (m0 < M) { C[c0] = acc00[r]; C[c0 + 16] = acc01[r]; }
        if (m1 < M) { C[c1] = acc10[r]; C[c1 + 16] = acc11[r]; }
    }
}

// ============================================================
// SSM scan: s_t = A @ s_{t-1} + bx_t   (per batch b, S=64)
// grid = B blocks, 64 threads; A row cached in registers; double-buffered LDS.
// ============================================================
__global__ void ssm_scan_kernel(const float* __restrict__ bx,
                                const float* __restrict__ A,
                                __bf16* __restrict__ states, int T) {
    const int b = blockIdx.x;
    const int i = threadIdx.x;              // 0..63
    float arow[64];
#pragma unroll
    for (int j = 0; j < 64; ++j) arow[j] = A[i * 64 + j];

    __shared__ float sbuf[2][64];
    sbuf[0][i] = 0.f;
    __syncthreads();

    const float* bxp = bx + (size_t)b * T * 64;
    __bf16*      stp = states + (size_t)b * T * 64;

    int cur = 0;
    for (int t = 0; t < T; ++t) {
        float acc = bxp[(size_t)t * 64 + i];
        const float* s = sbuf[cur];
#pragma unroll 8
        for (int j = 0; j < 64; ++j) acc += arow[j] * s[j];
        int nxt = cur ^ 1;
        sbuf[nxt][i] = acc;
        stp[(size_t)t * 64 + i] = (__bf16)acc;
        __syncthreads();
        cur = nxt;
    }
}

// ============================================================
// y = yc + D*x  -> LayerNorm(H=512) -> *g + b -> bf16 out
// grid = M rows, 512 threads.
// ============================================================
__global__ void add_d_ln_kernel(const float* __restrict__ yc,
                                const float* __restrict__ sig,
                                const float* __restrict__ Dv,
                                const float* __restrict__ g,
                                const float* __restrict__ bb,
                                __bf16* __restrict__ out) {
    const int row = blockIdx.x;
    const int h   = threadIdx.x;            // 512
    const size_t idx = (size_t)row * 512 + h;
    float y = yc[idx] + Dv[h] * sig[idx];

    __shared__ float red[512];
    red[h] = y;
    __syncthreads();
    for (int s = 256; s > 0; s >>= 1) {
        if (h < s) red[h] += red[h + s];
        __syncthreads();
    }
    float mu = red[0] * (1.f / 512.f);
    __syncthreads();
    float d = y - mu;
    red[h] = d * d;
    __syncthreads();
    for (int s = 256; s > 0; s >>= 1) {
        if (h < s) red[h] += red[h + s];
        __syncthreads();
    }
    float var = red[0] * (1.f / 512.f);
    float r = rsqrtf(var + 1e-5f);
    out[idx] = (__bf16)(d * r * g[h] + bb[h]);
}

// ============================================================
// pooled = mean_T(x); h = gelu(pooled@W1^T+b1); w = softmax(h@W2^T+b2)
// grid = B blocks, 256 threads.
// ============================================================
__global__ void pooled_attn_kernel(const float* __restrict__ x,
                                   const float* __restrict__ W1,
                                   const float* __restrict__ b1,
                                   const float* __restrict__ W2,
                                   const float* __restrict__ b2,
                                   float* __restrict__ watt, int T) {
    const int b = blockIdx.x;
    const int t = threadIdx.x;              // 256
    __shared__ float pooled[512];
    __shared__ float hbuf[256];
    __shared__ float logits[3];

    const float* xb = x + (size_t)b * T * 512;
    for (int h = t; h < 512; h += 256) {
        float s = 0.f;
        for (int tt = 0; tt < T; ++tt) s += xb[(size_t)tt * 512 + h];
        pooled[h] = s / (float)T;
    }
    __syncthreads();
    {
        float acc = b1[t];
        const float* w = W1 + (size_t)t * 512;
        for (int h = 0; h < 512; ++h) acc += w[h] * pooled[h];
        hbuf[t] = 0.5f * acc * (1.f + erff(acc * 0.70710678118654752f));
    }
    __syncthreads();
    if (t < 3) {
        float acc = b2[t];
        const float* w = W2 + (size_t)t * 256;
        for (int h = 0; h < 256; ++h) acc += w[h] * hbuf[h];
        logits[t] = acc;
    }
    __syncthreads();
    if (t == 0) {
        float m  = fmaxf(logits[0], fmaxf(logits[1], logits[2]));
        float e0 = expf(logits[0] - m), e1 = expf(logits[1] - m), e2 = expf(logits[2] - m);
        float s  = e0 + e1 + e2;
        watt[b * 3 + 0] = e0 / s;
        watt[b * 3 + 1] = e1 / s;
        watt[b * 3 + 2] = e2 / s;
    }
}

// ============================================================
// Final: linear-interp each coarse up_i to T, add up_b, attention-weighted sum.
// grid = B*T blocks, 512 threads.
// ============================================================
__global__ void combine_kernel(const float* __restrict__ up0,
                               const float* __restrict__ up1,
                               const float* __restrict__ up2,
                               const float* __restrict__ upb,
                               const float* __restrict__ watt,
                               float* __restrict__ out,
                               int T, int Ti0, int Ti1, int Ti2) {
    const int bt = blockIdx.x;
    const int h  = threadIdx.x;             // 512
    const int b  = bt / T;
    const int t  = bt % T;
    const float* ups[3] = {up0, up1, up2};
    const int    Tis[3] = {Ti0, Ti1, Ti2};

    float res = 0.f;
#pragma unroll
    for (int i = 0; i < 3; ++i) {
        const int Ti = Tis[i];
        float scale = (float)Ti / (float)T;
        float pos = ((float)t + 0.5f) * scale - 0.5f;
        pos = fminf(fmaxf(pos, 0.f), (float)(Ti - 1));
        int   i0 = (int)floorf(pos);
        int   i1 = (i0 + 1 < Ti) ? i0 + 1 : Ti - 1;
        float w  = pos - (float)i0;
        const float* u = ups[i];
        size_t base = (size_t)b * Ti * 512;
        float v = u[base + (size_t)i0 * 512 + h] * (1.f - w)
                + u[base + (size_t)i1 * 512 + h] * w
                + upb[i * 512 + h];
        res += watt[b * 3 + i] * v;
    }
    out[(size_t)bt * 512 + h] = res;
}

// ============================================================
// Host launcher
// ============================================================
extern "C" void kernel_launch(void* const* d_in, const int* in_sizes, int n_in,
                              void* d_out, int out_size, void* d_ws, size_t ws_size,
                              hipStream_t stream) {
    const float* x    = (const float*)d_in[0];
    const float* Am   = (const float*)d_in[1];
    const float* Bw   = (const float*)d_in[2];
    const float* Cw   = (const float*)d_in[3];
    const float* Dv   = (const float*)d_in[4];
    const float* ln_g = (const float*)d_in[5];
    const float* ln_b = (const float*)d_in[6];
    const float* upW  = (const float*)d_in[7];
    const float* upb  = (const float*)d_in[8];
    const float* aW1  = (const float*)d_in[9];
    const float* ab1  = (const float*)d_in[10];
    const float* aW2  = (const float*)d_in[11];
    const float* ab2  = (const float*)d_in[12];
    float* out = (float*)d_out;

    const int B = 4, T = 4096, H = 512, S = 64;
    const int T1 = (T + 7) / 2;   // 2051
    const int T2 = (T1 + 7) / 2;  // 1029
    const size_t N1 = (size_t)B * T1 * H;
    const size_t N2 = (size_t)B * T2 * H;

    // ---- carve workspace ----
    char* wsb = (char*)d_ws;
    size_t off = 0;
    auto alloc = [&](size_t bytes) -> void* {
        off = (off + 255) & ~(size_t)255;
        void* p = wsb + off;
        off += bytes;
        return p;
    };
    float*  cA1   = (float*)alloc(N1 * 4);
    float*  sigH  = (float*)alloc(N1 * 4);
    float*  sigM  = (float*)alloc(N2 * 4);
    float*  sigL  = (float*)alloc(N2 * 4);
    __bf16* sigbf = (__bf16*)alloc(N1 * 2);            // reused: sig-bf16, then ln-bf16
    float*  bxbuf = (float*)alloc((size_t)B * T1 * S * 4);
    __bf16* stbuf = (__bf16*)alloc((size_t)B * T1 * S * 2);
    float*  up0   = (float*)alloc(N1 * 4);             // doubles as yc buffer
    float*  up1   = (float*)alloc(N2 * 4);
    float*  up2   = (float*)alloc(N2 * 4);
    __bf16* Bwbf  = (__bf16*)alloc((size_t)3 * S * H * 2);
    __bf16* Cwbf  = (__bf16*)alloc((size_t)3 * H * S * 2);
    __bf16* upWbf = (__bf16*)alloc((size_t)3 * H * H * 2);
    float*  watt  = (float*)alloc(16 * 4);
    (void)ws_size; (void)n_in; (void)in_sizes; (void)out_size;

    auto cvt = [&](const float* src, __bf16* dst, size_t n) {
        int blocks = (int)((n + 255) / 256);
        f32_to_bf16_kernel<<<blocks, 256, 0, stream>>>(src, dst, n);
    };
    auto gemm = [&](const __bf16* A_, const __bf16* B_, float* C_, int M, int N, int K) {
        int tilesM = (M + 31) / 32;                    // 32 rows per wave
        dim3 grid(N / 32, (tilesM + 3) / 4);           // 4 waves per block
        gemm_bf16_wmma_kernel<<<grid, 128, 0, stream>>>(A_, B_, C_, M, N, K);
    };

    // ---- DWT (2 levels) ----
    dwt_kernel<<<B * T1, 512, 0, stream>>>(x,   cA1,  sigH, T,  T1);
    dwt_kernel<<<B * T2, 512, 0, stream>>>(cA1, sigL, sigM, T1, T2);

    // ---- weight conversions + attention weights (independent) ----
    cvt(Bw,  Bwbf,  (size_t)3 * S * H);
    cvt(Cw,  Cwbf,  (size_t)3 * H * S);
    cvt(upW, upWbf, (size_t)3 * H * H);
    pooled_attn_kernel<<<B, 256, 0, stream>>>(x, aW1, ab1, aW2, ab2, watt, T);

    // ---- per-signal SSM pipeline (order: high, mid, low) ----
    const float* sigs[3] = {sigH, sigM, sigL};
    const int    Tis[3]  = {T1, T2, T2};
    float*       ups[3]  = {up0, up1, up2};

    for (int i = 0; i < 3; ++i) {
        const float* sig = sigs[i];
        const int Ti = Tis[i];
        const int M  = B * Ti;
        float* upi = ups[i];                 // also serves as yc scratch

        cvt(sig, sigbf, (size_t)M * H);
        // bx = sig @ Bw_i^T : (M,H)x(S,H)^T -> (M,S)
        gemm(sigbf, Bwbf + (size_t)i * S * H, bxbuf, M, S, H);
        // sequential state-space scan (fp32), emits bf16 states
        ssm_scan_kernel<<<B, 64, 0, stream>>>(bxbuf, Am + (size_t)i * S * S, stbuf, Ti);
        // yc = states @ Cw_i^T : (M,S)x(H,S)^T -> (M,H)
        gemm(stbuf, Cwbf + (size_t)i * H * S, upi, M, H, S);
        // y = yc + D*x, LayerNorm, -> bf16
        add_d_ln_kernel<<<M, 512, 0, stream>>>(upi, sig, Dv + (size_t)i * H,
                                               ln_g + (size_t)i * H,
                                               ln_b + (size_t)i * H, sigbf);
        // up_i = ln @ upW_i^T at coarse resolution (interp commutes with matmul)
        gemm(sigbf, upWbf + (size_t)i * H * H, upi, M, H, H);
    }

    // ---- interp to T, bias, attention-weighted sum ----
    combine_kernel<<<B * T, 512, 0, stream>>>(up0, up1, up2, upb, watt, out,
                                              T, T1, T2, T2);
}
